// VideoColBERT_19232863552031
// MI455X (gfx1250) — matrix operand order repaired and checked by reading, not compiled, not run on hardware
//
#include <hip/hip_runtime.h>

// ---------------------------------------------------------------------------
// VideoColBERT on MI455X (gfx1250): f16 WMMA GEMMs + f32 accum / LN / softmax
// GEMM uses 32x256 block tiles, 16x64 wave tiles, double-buffered LDS with
// GLOBAL_LOAD_ASYNC_TO_LDS (ASYNCcnt) when the toolchain exposes the builtin.
// ---------------------------------------------------------------------------

#define B_    256
#define Fr    30
#define Dm    512
#define Hh    8
#define DHd   64
#define FFd   2048
#define Ssq   32
#define Ll    4
#define Mrows (B_ * Ssq)          // 8192 activation rows
#define LDH_  72                  // 64 data halves + 8 pad (144B rows, 16B aligned)

typedef __attribute__((ext_vector_type(16))) _Float16 v16h;
typedef __attribute__((ext_vector_type(8)))  _Float16 v8h;
typedef __attribute__((ext_vector_type(8)))  float    v8f;
typedef __attribute__((ext_vector_type(4)))  int      v4i;

union FragH { v16h v; v8h h[2]; };
union AccF  { v8f  v; float f[8]; };

#if defined(__has_builtin)
#if __has_builtin(__builtin_amdgcn_global_load_async_to_lds_b128)
#define HAVE_ASYNC_COPY 1
#endif
#endif
#ifndef HAVE_ASYNC_COPY
#define HAVE_ASYNC_COPY 0
#endif

#if HAVE_ASYNC_COPY
// Builtin signature (from hipcc diagnostic): (int4 AS1*, int4 AS3*, Imm, Imm)
typedef __attribute__((address_space(1))) v4i* gptr_t;
typedef __attribute__((address_space(3))) v4i* lptr_t;
#endif

__device__ __forceinline__ void wait_async0() {
#if HAVE_ASYNC_COPY
#if __has_builtin(__builtin_amdgcn_s_wait_asynccnt)
    __builtin_amdgcn_s_wait_asynccnt(0);
#else
    asm volatile("s_wait_asynccnt 0x0" ::: "memory");
#endif
#endif
}

// A-matrix fragment (16x32 f16): lane l<16 -> row=l, K = {0..7, 16..23};
// lane l>=16 -> row=l-16, K = {8..15, 24..31}   (ISA 7.12.2 dense A table)
__device__ __forceinline__ v16h frag_a(const _Float16* base, int row0, int ldh, int kk) {
    const int lane  = threadIdx.x & 31;
    const int r     = row0 + (lane & 15);
    const int khalf = lane >> 4;
    const _Float16* p = base + r * ldh + kk + khalf * 8;
    FragH f;
    f.h[0] = *(const v8h*)(p);
    f.h[1] = *(const v8h*)(p + 16);
    return f.v;
}

// B-matrix fragment (32x16 f16), stored row-major [n][k]:
// lane l<16 -> col=l, K = {0..7, 16..23}; lane l>=16 -> col=l-16, K = {8..15, 24..31}
// (same K order as the A fragment)
__device__ __forceinline__ v16h frag_b(const _Float16* base, int row0, int ldh, int kk) {
    const int lane  = threadIdx.x & 31;
    const int r     = row0 + (lane & 15);
    const int khalf = lane >> 4;
    const _Float16* p = base + r * ldh + kk + khalf * 8;
    FragH f;
    f.h[0] = *(const v8h*)(p);
    f.h[1] = *(const v8h*)(p + 16);
    return f.v;
}

#define WMMA_F16(a, b, c) \
    __builtin_amdgcn_wmma_f32_16x16x32_f16(false, (a), false, (b), (short)0, (c), false, false)

// Stage one K-block: A tile 32x64 halves, W tile 256x64 halves.
// 256 threads: thread t -> row t/8 (+i*32 for W), 16B column chunk (t%8)*8.
__device__ __forceinline__ void stage_tile(
    const _Float16* __restrict__ A, const _Float16* __restrict__ W,
    _Float16* As, _Float16* Ws, int m0, int n0, int K, int k0, int ar, int ac)
{
#if HAVE_ASYNC_COPY
    __builtin_amdgcn_global_load_async_to_lds_b128(
        (gptr_t)&A[(long)(m0 + ar) * K + k0 + ac],
        (lptr_t)&As[ar * LDH_ + ac], 0, 0);
#pragma unroll
    for (int i = 0; i < 8; ++i) {
        const int r = ar + i * 32;
        __builtin_amdgcn_global_load_async_to_lds_b128(
            (gptr_t)&W[(long)(n0 + r) * K + k0 + ac],
            (lptr_t)&Ws[r * LDH_ + ac], 0, 0);
    }
#else
    *(v8h*)&As[ar * LDH_ + ac] = *(const v8h*)&A[(long)(m0 + ar) * K + k0 + ac];
#pragma unroll
    for (int i = 0; i < 8; ++i) {
        const int r = ar + i * 32;
        *(v8h*)&Ws[r * LDH_ + ac] = *(const v8h*)&W[(long)(n0 + r) * K + k0 + ac];
    }
#endif
}

// ---------------------------------------------------------------------------
// Generic GEMM: C[M][N] = A[M][K] * W[N][K]^T (+bias) (+ReLU), f16 in, f16/f32 out
// Block: 256 threads = 8 waves; C tile 32x256; wave tile 16x64; K-block 64,
// double-buffered LDS with async-copy overlap.
// Requires M%32==0, N%256==0, K%64==0 (all call sites satisfy this).
// ---------------------------------------------------------------------------
template<bool RELU, bool OUTF16>
__global__ __launch_bounds__(256) void gemm_wmma(
    const _Float16* __restrict__ A,
    const _Float16* __restrict__ W,
    const float*    __restrict__ bias,
    void*           __restrict__ Cout,
    int M, int N, int K)
{
    __shared__ _Float16 As[2][32 * LDH_];
    __shared__ _Float16 Ws[2][256 * LDH_];

    const int m0 = blockIdx.y * 32;
    const int n0 = blockIdx.x * 256;

    const int t    = threadIdx.x;
    const int lane = t & 31;
    const int w    = t >> 5;
    const int wr   = w >> 2;                 // 0..1 : 16-row tile
    const int wc   = w & 3;                  // 0..3 : 64-col tile

    const int ar = t >> 3;                   // staging row
    const int ac = (t & 7) * 8;              // staging col (halves)

    v8f acc[4] = {};

    stage_tile(A, W, As[0], Ws[0], m0, n0, K, 0, ar, ac);
    wait_async0();
    __syncthreads();

    const int ntiles = K >> 6;
    for (int i = 0; i < ntiles; ++i) {
        const int cur = i & 1;
        if (i + 1 < ntiles)   // prefetch next K-block into the other buffer
            stage_tile(A, W, As[cur ^ 1], Ws[cur ^ 1], m0, n0, K, (i + 1) << 6, ar, ac);

#pragma unroll
        for (int kk = 0; kk < 64; kk += 32) {
            v16h a = frag_a(As[cur], wr * 16, LDH_, kk);
#pragma unroll
            for (int ni = 0; ni < 4; ++ni) {
                v16h b = frag_b(Ws[cur], wc * 64 + ni * 16, LDH_, kk);
                acc[ni] = WMMA_F16(a, b, acc[ni]);
            }
        }
        wait_async0();        // next buffer landed (overlapped with WMMAs above)
        __syncthreads();      // everyone done reading cur buffer
    }

    // Epilogue. C/D layout: VGPR j, lanes 0-15 -> (m=j, n=lane); lanes 16-31 -> (m=j+8, n=lane-16)
    const int khalf = lane >> 4;
    const int mbase = m0 + wr * 16 + khalf * 8;
#pragma unroll
    for (int ni = 0; ni < 4; ++ni) {
        const int nl = n0 + wc * 64 + ni * 16 + (lane & 15);
        const float bv = bias ? bias[nl] : 0.f;
        AccF u; u.v = acc[ni];
#pragma unroll
        for (int j = 0; j < 8; ++j) {
            float c = u.f[j] + bv;
            if (RELU) c = fmaxf(c, 0.f);
            const long row = mbase + j;
            if (OUTF16) ((_Float16*)Cout)[row * N + nl] = (_Float16)c;
            else        ((float*)Cout)[row * N + nl] = c;
        }
    }
}

// ---------------------------------------------------------------------------
// Attention: one wave per (b, h). S=32, dh=64.  scores = Q K^T / 8 -> softmax
// -> O = P V.  Q,K row-major in LDS are directly WMMA A/B operands; V is
// transposed through LDS for the P*V B operand.
// ---------------------------------------------------------------------------
__global__ __launch_bounds__(32) void attn_kernel(
    const _Float16* __restrict__ qkv,      // [B][32][1536]
    _Float16*       __restrict__ attn_out) // [B][32][512]
{
    const int LDQ = 72;   // 64 + 8 pad halves
    const int LDV = 40;   // 32 + 8 pad halves (80B rows, 16B aligned)
    const int LDP = 40;
    __shared__ _Float16 Qs[32 * LDQ];
    __shared__ _Float16 Ks[32 * LDQ];
    __shared__ _Float16 Vt[64 * LDV];      // V^T : [dh][s]
    __shared__ float    Pf[32 * 33];       // f32 scores
    __shared__ _Float16 Ph[32 * LDP];      // f16 softmax probs

    const int h    = blockIdx.x;
    const int b    = blockIdx.y;
    const int lane = threadIdx.x;          // == row s

    const _Float16* base = qkv + (long)b * Ssq * (3 * Dm);
    {
        const _Float16* qrow = base + lane * (3 * Dm) + h * DHd;
        const _Float16* krow = base + lane * (3 * Dm) + Dm + h * DHd;
        const _Float16* vrow = base + lane * (3 * Dm) + 2 * Dm + h * DHd;
#pragma unroll
        for (int c = 0; c < 64; c += 8) {
            *(v8h*)&Qs[lane * LDQ + c] = *(const v8h*)&qrow[c];
            *(v8h*)&Ks[lane * LDQ + c] = *(const v8h*)&krow[c];
        }
#pragma unroll
        for (int d = 0; d < 64; ++d) Vt[d * LDV + lane] = vrow[d];
    }
    __syncthreads();

    // scores (2x2 tiles of 16x16), K = dh = 64 -> 2 WMMA each
#pragma unroll
    for (int ti = 0; ti < 2; ++ti) {
#pragma unroll
        for (int tj = 0; tj < 2; ++tj) {
            v8f acc = {};
#pragma unroll
            for (int kk = 0; kk < 64; kk += 32) {
                v16h a  = frag_a(Qs, ti * 16, LDQ, kk);
                v16h bb = frag_b(Ks, tj * 16, LDQ, kk);
                acc = WMMA_F16(a, bb, acc);
            }
            AccF u; u.v = acc;
            const int khalf = lane >> 4;
            const int n  = tj * 16 + (lane & 15);
            const int mb = ti * 16 + khalf * 8;
#pragma unroll
            for (int j = 0; j < 8; ++j) Pf[(mb + j) * 33 + n] = u.f[j] * 0.125f;
        }
    }
    __syncthreads();

    // softmax: lane == row
    {
        float vals[32];
        float mx = -1e30f;
#pragma unroll
        for (int i = 0; i < 32; ++i) { vals[i] = Pf[lane * 33 + i]; mx = fmaxf(mx, vals[i]); }
        float s = 0.f;
#pragma unroll
        for (int i = 0; i < 32; ++i) { vals[i] = __expf(vals[i] - mx); s += vals[i]; }
        const float inv = 1.f / s;
#pragma unroll
        for (int i = 0; i < 32; ++i) Ph[lane * LDP + i] = (_Float16)(vals[i] * inv);
    }
    __syncthreads();

    // O = P @ V : 2 x 4 tiles, K = 32 -> single WMMA each
#pragma unroll
    for (int mi = 0; mi < 2; ++mi) {
        v16h a = frag_a(Ph, mi * 16, LDP, 0);
#pragma unroll
        for (int ni = 0; ni < 4; ++ni) {
            v16h bb = frag_b(Vt, ni * 16, LDV, 0);
            v8f acc = {};
            acc = WMMA_F16(a, bb, acc);
            AccF u; u.v = acc;
            const int khalf = lane >> 4;
            const int n  = ni * 16 + (lane & 15);
            const int mb = mi * 16 + khalf * 8;
#pragma unroll
            for (int j = 0; j < 8; ++j)
                attn_out[((long)b * Ssq + mb + j) * Dm + h * DHd + n] = (_Float16)u.f[j];
        }
    }
}

// ---------------------------------------------------------------------------
// Fused residual + LayerNorm per row of 512; writes f32 x and f16 shadow.
// ---------------------------------------------------------------------------
__global__ __launch_bounds__(256) void ln_residual(
    float* __restrict__ x, const float* __restrict__ y,
    const float* __restrict__ g, const float* __restrict__ beta,
    _Float16* __restrict__ xh)
{
    const int row = blockIdx.x;
    const int t   = threadIdx.x;
    __shared__ float s1[256], s2[256];

    const long base = (long)row * Dm;
    float v0 = x[base + t]       + y[base + t];
    float v1 = x[base + 256 + t] + y[base + 256 + t];
    s1[t] = v0 + v1;
    s2[t] = v0 * v0 + v1 * v1;
    __syncthreads();
    for (int off = 128; off > 0; off >>= 1) {
        if (t < off) { s1[t] += s1[t + off]; s2[t] += s2[t + off]; }
        __syncthreads();
    }
    const float mean = s1[0] * (1.f / Dm);
    const float var  = s2[0] * (1.f / Dm) - mean * mean;
    const float rstd = rsqrtf(var + 1e-5f);
    const float o0 = (v0 - mean) * rstd * g[t]       + beta[t];
    const float o1 = (v1 - mean) * rstd * g[256 + t] + beta[256 + t];
    x[base + t] = o0;            x[base + 256 + t] = o1;
    xh[base + t] = (_Float16)o0; xh[base + 256 + t] = (_Float16)o1;
}

// L2-normalize rows of 512, f32 in -> f16 out
__global__ __launch_bounds__(256) void l2norm_rows(
    const float* __restrict__ in, _Float16* __restrict__ out)
{
    const int row = blockIdx.x;
    const int t   = threadIdx.x;
    __shared__ float s[256];
    const long base = (long)row * Dm;
    const float v0 = in[base + t], v1 = in[base + 256 + t];
    s[t] = v0 * v0 + v1 * v1;
    __syncthreads();
    for (int off = 128; off > 0; off >>= 1) {
        if (t < off) s[t] += s[t + off];
        __syncthreads();
    }
    const float inv = 1.f / fmaxf(sqrtf(s[0]), 1e-12f);
    out[base + t]       = (_Float16)(v0 * inv);
    out[base + 256 + t] = (_Float16)(v1 * inv);
}

// Build x0 = [frames ; expansion tokens] (f32 + f16 shadow)
__global__ __launch_bounds__(256) void prep_x(
    const float* __restrict__ frames, const float* __restrict__ exptok,
    float* __restrict__ xf, _Float16* __restrict__ xh)
{
    const int i  = blockIdx.x * 256 + threadIdx.x;     // < 8192*512
    const int d  = i & (Dm - 1);
    const int bs = i >> 9;
    const int s  = bs & (Ssq - 1);
    const int b  = bs >> 5;
    const float v = (s < Fr) ? frames[((long)b * Fr + s) * Dm + d]
                             : exptok[(s - Fr) * Dm + d];
    xf[i] = v;
    xh[i] = (_Float16)v;
}

__global__ __launch_bounds__(256) void f32_to_f16(
    const float* __restrict__ in, _Float16* __restrict__ out, int n)
{
    const int i = blockIdx.x * 256 + threadIdx.x;
    if (i < n) out[i] = (_Float16)in[i];
}

// MaxSim reduction + output assembly: out = [sim_f+sim_v | sim_f | sim_v]
__global__ __launch_bounds__(256) void colbert_reduce(
    const float* __restrict__ Sf,   // [256][7680]  (c*30+f)
    const float* __restrict__ Sv,   // [256][8192]  (c*32+v)
    float* __restrict__ out)
{
    const int i = blockIdx.x * 256 + threadIdx.x;      // < 65536
    const int b = i >> 8, c = i & 255;
    const float* pf = Sf + (long)b * (B_ * Fr) + c * Fr;
    float mf = pf[0];
#pragma unroll
    for (int f = 1; f < Fr; ++f) mf = fmaxf(mf, pf[f]);
    const float* pv = Sv + (long)b * (B_ * Ssq) + c * Ssq;
    float mv = pv[0];
#pragma unroll
    for (int v = 1; v < Ssq; ++v) mv = fmaxf(mv, pv[v]);
    out[i]          = mf + mv;
    out[65536 + i]  = mf;
    out[131072 + i] = mv;
}

// ---------------------------------------------------------------------------
// Host orchestration
// ---------------------------------------------------------------------------
extern "C" void kernel_launch(void* const* d_in, const int* in_sizes, int n_in,
                              void* d_out, int out_size, void* d_ws, size_t ws_size,
                              hipStream_t stream) {
    const float* text   = (const float*)d_in[0];
    const float* frames = (const float*)d_in[1];
    const float* exptok = (const float*)d_in[2];
    const float* Wqkv   = (const float*)d_in[3];
    const float* bqkv   = (const float*)d_in[4];
    const float* Wo     = (const float*)d_in[5];
    const float* bo     = (const float*)d_in[6];
    const float* ln1w   = (const float*)d_in[7];
    const float* ln1b   = (const float*)d_in[8];
    const float* W1     = (const float*)d_in[9];
    const float* b1     = (const float*)d_in[10];
    const float* W2     = (const float*)d_in[11];
    const float* b2     = (const float*)d_in[12];
    const float* ln2w   = (const float*)d_in[13];
    const float* ln2b   = (const float*)d_in[14];
    float* out = (float*)d_out;

    char*  ws  = (char*)d_ws;
    size_t off = 0;
    auto take = [&](size_t bytes) -> char* {
        char* p = ws + off;
        off += (bytes + 255) & ~(size_t)255;
        return p;
    };

    float*    xf    = (float*)   take((size_t)Mrows * Dm * 4);
    _Float16* xh    = (_Float16*)take((size_t)Mrows * Dm * 2);
    _Float16* big   = (_Float16*)take((size_t)Mrows * FFd * 2);  // qkv (x1536) / ff hidden (x2048)
    _Float16* attnh = (_Float16*)take((size_t)Mrows * Dm * 2);
    float*    yf    = (float*)   take((size_t)Mrows * Dm * 4);
    _Float16* whqkv = (_Float16*)take((size_t)Ll * 3 * Dm * Dm * 2);
    _Float16* who   = (_Float16*)take((size_t)Ll * Dm * Dm * 2);
    _Float16* wh1   = (_Float16*)take((size_t)Ll * FFd * Dm * 2);
    _Float16* wh2   = (_Float16*)take((size_t)Ll * Dm * FFd * 2);
    _Float16* th    = (_Float16*)take((size_t)B_ * Dm * 2);
    _Float16* fn    = (_Float16*)take((size_t)B_ * Fr * Dm * 2);
    _Float16* vn    = (_Float16*)take((size_t)Mrows * Dm * 2);
    float*    Sv    = (float*)   take((size_t)B_ * Mrows * 4);
    float*    Sf    = (float*)   take((size_t)B_ * B_ * Fr * 4);

    // 1) weights -> f16
    const int nqkv = Ll * 3 * Dm * Dm, no_ = Ll * Dm * Dm, n1 = Ll * FFd * Dm, n2 = Ll * Dm * FFd;
    f32_to_f16<<<(nqkv + 255) / 256, 256, 0, stream>>>(Wqkv, whqkv, nqkv);
    f32_to_f16<<<(no_  + 255) / 256, 256, 0, stream>>>(Wo,   who,   no_);
    f32_to_f16<<<(n1   + 255) / 256, 256, 0, stream>>>(W1,   wh1,   n1);
    f32_to_f16<<<(n2   + 255) / 256, 256, 0, stream>>>(W2,   wh2,   n2);

    // 2) x0 = [frames ; expansion]
    prep_x<<<(Mrows * Dm) / 256, 256, 0, stream>>>(frames, exptok, xf, xh);

    // 3) transformer layers
    for (int l = 0; l < Ll; ++l) {
        gemm_wmma<false, true><<<dim3((3 * Dm) / 256, Mrows / 32), 256, 0, stream>>>(
            xh, whqkv + (size_t)l * 3 * Dm * Dm, bqkv + l * 3 * Dm, big, Mrows, 3 * Dm, Dm);

        attn_kernel<<<dim3(Hh, B_), 32, 0, stream>>>(big, attnh);

        gemm_wmma<false, false><<<dim3(Dm / 256, Mrows / 32), 256, 0, stream>>>(
            attnh, who + (size_t)l * Dm * Dm, bo + l * Dm, yf, Mrows, Dm, Dm);

        ln_residual<<<Mrows, 256, 0, stream>>>(xf, yf, ln1w + l * Dm, ln1b + l * Dm, xh);

        gemm_wmma<true, true><<<dim3(FFd / 256, Mrows / 32), 256, 0, stream>>>(
            xh, wh1 + (size_t)l * FFd * Dm, b1 + l * FFd, big, Mrows, FFd, Dm);

        gemm_wmma<false, false><<<dim3(Dm / 256, Mrows / 32), 256, 0, stream>>>(
            big, wh2 + (size_t)l * Dm * FFd, b2 + l * Dm, yf, Mrows, Dm, FFd);

        ln_residual<<<Mrows, 256, 0, stream>>>(xf, yf, ln2w + l * Dm, ln2b + l * Dm, xh);
    }

    // 4) normalized embeddings
    l2norm_rows<<<B_,      256, 0, stream>>>(text,   th);
    l2norm_rows<<<B_ * Fr, 256, 0, stream>>>(frames, fn);
    l2norm_rows<<<Mrows,   256, 0, stream>>>(xf,     vn);

    // 5) similarity GEMMs (no bias)
    gemm_wmma<false, false><<<dim3(Mrows / 256, B_ / 32), 256, 0, stream>>>(
        th, vn, nullptr, Sv, B_, Mrows, Dm);
    gemm_wmma<false, false><<<dim3((B_ * Fr) / 256, B_ / 32), 256, 0, stream>>>(
        th, fn, nullptr, Sf, B_, B_ * Fr, Dm);

    // 6) MaxSim reduction -> (sum, sim_f, sim_v)
    colbert_reduce<<<(B_ * B_) / 256, 256, 0, stream>>>(Sf, Sv, out);
}
